// MultiHeadAttention_22170621182614
// MI455X (gfx1250) — matrix operand order repaired and checked by reading, not compiled, not run on hardware
//
#include <hip/hip_runtime.h>
#include <hip/hip_bf16.h>
#include <stdint.h>

// ---------------------------------------------------------------------------
// MI455X (gfx1250) relative-position MHA.  wave32, bf16 WMMA 16x16x32, f32 acc.
// Shapes: B=8, T=512, D=512, NH=8, DH=64.
// rel_emb (537 MB) is streamed exactly once through LDS with
// global_load_async_to_lds_b128 double buffering (ASYNCcnt-tracked).
// ---------------------------------------------------------------------------

#define Bn  8
#define Tn  512
#define Dm  512
#define NHn 8
#define DHn 64
#define TT  8      // t-values per k_pos block (double-buffered async pipeline)

typedef __attribute__((ext_vector_type(16))) __bf16 bf16x16;
typedef __attribute__((ext_vector_type(8)))  float  f32x8;

union FragB16 {
    bf16x16  v;
    uint32_t u[8];
};

__device__ __forceinline__ uint32_t f2bf(float f) {
    union { float f; uint32_t u; } x; x.f = f;
    uint32_t u = x.u;
    return (u + 0x7FFFu + ((u >> 16) & 1u)) >> 16;   // RNE truncate to bf16 bits
}

// pack two f32 into bf16x2 bits; prefer native v_cvt_pk_bf16_f32 when declared
__device__ __forceinline__ uint32_t pkbf(float lo, float hi) {
#if __has_builtin(__builtin_amdgcn_cvt_pk_bf16_f32)
    typedef __attribute__((ext_vector_type(2))) __bf16 bf16x2;
    union { bf16x2 v; uint32_t u; } x;
    x.v = __builtin_amdgcn_cvt_pk_bf16_f32(lo, hi);
    return x.u;
#else
    return f2bf(lo) | (f2bf(hi) << 16);
#endif
}

// D = A x B + C, bf16 inputs, f32 accumulate
__device__ __forceinline__ f32x8 wmma_bf16(bf16x16 a, bf16x16 b, f32x8 c) {
    return __builtin_amdgcn_wmma_f32_16x16x32_bf16(false, a, false, b,
                                                   (short)0, c, false, false);
}

// ---- CDNA5 async global->LDS (ASYNCcnt) ------------------------------------
// the LDS destination comes from the vdst VGPR; the low 32 bits of a generic LDS pointer are
// the wave-relative LDS byte offset (flat aperture mapping, ISA 10.2).
__device__ __forceinline__ void async_load_b128(const float* g, float* l) {
    uint32_t loff = (uint32_t)(uintptr_t)l;
    asm volatile("global_load_async_to_lds_b128 %0, %1, off"
                 :: "v"(loff), "v"(g)
                 : "memory");
}

__device__ __forceinline__ void wait_asynccnt0() {
#if __has_builtin(__builtin_amdgcn_s_wait_asynccnt)
    __builtin_amdgcn_s_wait_asynccnt(0);
#else
    asm volatile("s_wait_asynccnt 0x0" ::: "memory");
#endif
}

// ---- Fragment loaders (per CDNA5 ISA 7.12.2 VGPR layouts, wave32) ----------
// A (16x32 bf16): lane m = lane&15; lanes<16 hold K 0..7 & 16..23, lanes>=16
// hold K 8..15 & 24..31 (packed pairs per VGPR).
__device__ __forceinline__ bf16x16 load_a_bf16(const unsigned short* base,
                                               int ldRow, int lane) {
    int m  = lane & 15;
    int kh = (lane >> 4) << 3;            // 0 or 8
    const unsigned short* p = base + (size_t)m * ldRow + kh;
    uint4 lo = *(const uint4*)p;          // k = kh .. kh+7
    uint4 hi = *(const uint4*)(p + 16);   // k = kh+16 .. kh+23
    FragB16 r;
    r.u[0] = lo.x; r.u[1] = lo.y; r.u[2] = lo.z; r.u[3] = lo.w;
    r.u[4] = hi.x; r.u[5] = hi.y; r.u[6] = hi.z; r.u[7] = hi.w;
    return r.v;
}

// A with only first 8 rows valid (batch-on-M trick); rows 8..15 = 0.
__device__ __forceinline__ bf16x16 load_a_bf16_rows8(const unsigned short* base,
                                                     size_t ldRow, int lane) {
    int m  = lane & 15;
    int kh = (lane >> 4) << 3;
    FragB16 r;
    if (m < 8) {
        const unsigned short* p = base + (size_t)m * ldRow + kh;
        uint4 lo = *(const uint4*)p;
        uint4 hi = *(const uint4*)(p + 16);
        r.u[0] = lo.x; r.u[1] = lo.y; r.u[2] = lo.z; r.u[3] = lo.w;
        r.u[4] = hi.x; r.u[5] = hi.y; r.u[6] = hi.z; r.u[7] = hi.w;
    } else {
        #pragma unroll
        for (int j = 0; j < 8; ++j) r.u[j] = 0u;
    }
    return r.v;
}

// A from f32 source (scores -> P), converting to bf16 in-register.
__device__ __forceinline__ bf16x16 load_a_f32(const float* base,
                                              int ldRow, int lane) {
    int m  = lane & 15;
    int kh = (lane >> 4) << 3;
    const float* p = base + (size_t)m * ldRow + kh;
    FragB16 r;
    #pragma unroll
    for (int j = 0; j < 4; ++j) {
        float2 t = *(const float2*)(p + 2 * j);
        r.u[j] = pkbf(t.x, t.y);
    }
    #pragma unroll
    for (int j = 0; j < 4; ++j) {
        float2 t = *(const float2*)(p + 16 + 2 * j);
        r.u[4 + j] = pkbf(t.x, t.y);
    }
    return r.v;
}

// B (32x16 bf16): column n = row n of source (K contiguous). Lanes<16 hold
// K 0..7 & 16..23, lanes>=16 hold K 8..15 & 24..31, packed pairs per VGPR.
__device__ __forceinline__ bf16x16 load_b_bf16(const unsigned short* base,
                                               int ldRow, int lane) {
    int n  = lane & 15;
    int kh = (lane >> 4) << 3;            // 0 or 8
    const unsigned short* p = base + (size_t)n * ldRow + kh;
    uint4 lo = *(const uint4*)p;          // k = kh .. kh+7
    uint4 hi = *(const uint4*)(p + 16);   // k = kh+16 .. kh+23
    FragB16 r;
    r.u[0] = lo.x; r.u[1] = lo.y; r.u[2] = lo.z; r.u[3] = lo.w;
    r.u[4] = hi.x; r.u[5] = hi.y; r.u[6] = hi.z; r.u[7] = hi.w;
    return r.v;
}

// B from f32 source (global or LDS), converting to bf16 in-register.
__device__ __forceinline__ bf16x16 load_b_f32(const float* base,
                                              int ldRow, int lane) {
    int n  = lane & 15;
    int kh = (lane >> 4) << 3;
    const float* p = base + (size_t)n * ldRow + kh;
    FragB16 r;
    #pragma unroll
    for (int j = 0; j < 8; ++j) {
        float2 t = *(const float2*)(p + 2 * j + (j >> 2) * 8);
        r.u[j] = pkbf(t.x, t.y);
    }
    return r.v;
}

// ---------------------------------------------------------------------------
// K0: f32 -> bf16 bulk convert (x and the four weight matrices)
// ---------------------------------------------------------------------------
__global__ __launch_bounds__(256)
void k_cvt_bf16(const float* __restrict__ src, unsigned short* __restrict__ dst, int n) {
    int i = blockIdx.x * 256 + threadIdx.x;
    if (i < n) dst[i] = (unsigned short)f2bf(src[i]);
}

// ---------------------------------------------------------------------------
// K1: QKV projection.  out = x @ W^T + b, reshaped per-head.
//   which 0 -> Q[b][h][t][d], 1 -> K[b][h][t][d], 2 -> V^T[b][h][d][t]
// grid (256 m-tiles, 8 heads), block 128 (4 waves; wave w -> 16-col subtile)
// ---------------------------------------------------------------------------
__global__ __launch_bounds__(128)
void k_qkv(const unsigned short* __restrict__ xb,
           const unsigned short* __restrict__ w,
           const float* __restrict__ bias,
           unsigned short* __restrict__ out, int which) {
    int lane = threadIdx.x & 31, wv = threadIdx.x >> 5;
    int mt = blockIdx.x;              // row tile (16 rows of (b,t))
    int h  = blockIdx.y;              // head == 64-col block
    int c0 = h * 64 + wv * 16;        // output column base
    const unsigned short* abase = xb + (size_t)mt * 16 * Dm;
    const unsigned short* bbase = w + (size_t)c0 * Dm;
    f32x8 c = {};
    #pragma unroll 4
    for (int k = 0; k < Dm; k += 32)
        c = wmma_bf16(load_a_bf16(abase + k, Dm, lane),
                      load_b_bf16(bbase + k, Dm, lane), c);

    int col = lane & 15, m0 = (lane >> 4) << 3;
    float bvv = bias[c0 + col];
    int dcol = wv * 16 + col;         // d within head
    #pragma unroll
    for (int r = 0; r < 8; ++r) {
        int g  = mt * 16 + m0 + r;    // flattened (b*T + t)
        int bb = g >> 9, t = g & (Tn - 1);
        unsigned short val = (unsigned short)f2bf(c[r] + bvv);
        if (which == 2)   // V transposed: [b][h][d][t]
            out[(((size_t)(bb * NHn + h) * DHn + dcol) << 9) + t] = val;
        else              // Q/K: [b][h][t][d]
            out[((size_t)((bb * NHn + h) << 9 | t)) * DHn + dcol] = val;
    }
}

// ---------------------------------------------------------------------------
// K2: position scores.  For fixed (h,t): pos[b,s] = sum_d Q[b,h,t,d]*rel[t,s,h,d]
// Batch (8) rides the WMMA M axis (rows 8..15 masked).  Each block owns
// (h, s-block of 64, TT consecutive t) and pipelines the rel stream through
// LDS with async b128 loads, double buffered: prefetch tile t+1 while the
// WMMAs consume tile t.  Writes scores[b][h][t][s].
// grid (T/TT, 8, NH), block 128; LDS 2 x 16 KB.
// ---------------------------------------------------------------------------
__global__ __launch_bounds__(128)
void k_pos(const unsigned short* __restrict__ qb,
           const float* __restrict__ rel,
           float* __restrict__ scores) {
    __shared__ float tile[2][64 * 64];   // [buf][s(64) x d(64)] f32
    int tid  = threadIdx.x;
    int lane = tid & 31, wv = tid >> 5;
    int h  = blockIdx.z;
    int s0 = blockIdx.y * 64;
    int t0 = blockIdx.x * TT;
    const size_t bstride = (size_t)NHn * Tn * DHn;   // Q batch stride

    // prefetch tile for t0 into buf 0: 64 rows x 256 B, 8 b128 per thread
    {
        const float* gsrc = rel + ((size_t)t0 * Tn + s0) * Dm + h * DHn;
        #pragma unroll
        for (int j = 0; j < 8; ++j) {
            int lin = tid + j * 128;
            int row = lin >> 4, c4 = (lin & 15) << 2;
            async_load_b128(gsrc + (size_t)row * Dm + c4, &tile[0][row * 64 + c4]);
        }
    }

    for (int i = 0; i < TT; ++i) {
        int t = t0 + i;
        wait_asynccnt0();      // own async tile writes landed in LDS
        __syncthreads();       // publish across the 4 waves

        if (i + 1 < TT) {      // prefetch next t into the other buffer
            const float* gsrc = rel + ((size_t)(t + 1) * Tn + s0) * Dm + h * DHn;
            float* dst = &tile[(i + 1) & 1][0];
            #pragma unroll
            for (int j = 0; j < 8; ++j) {
                int lin = tid + j * 128;
                int row = lin >> 4, c4 = (lin & 15) << 2;
                async_load_b128(gsrc + (size_t)row * Dm + c4, dst + row * 64 + c4);
            }
        }

        // consume current buffer: wave wv handles s-tile [s0+wv*16, +16)
        const float* lb = &tile[i & 1][(wv * 16) * 64];
        const unsigned short* qbase = qb + (size_t)(h * Tn + t) * DHn;
        f32x8 c = {};
        #pragma unroll
        for (int k = 0; k < DHn; k += 32)
            c = wmma_bf16(load_a_bf16_rows8(qbase + k, bstride, lane),
                          load_b_f32(lb + k, 64, lane), c);

        if (lane < 16) {       // rows 0..7 == b = 0..7
            int scol = s0 + wv * 16 + lane;
            #pragma unroll
            for (int r = 0; r < 8; ++r)
                scores[(((size_t)(r * NHn + h) * Tn + t) << 9) + scol] = c[r];
        }
    }
}

// ---------------------------------------------------------------------------
// K3: content scores + combine:  scores = (Q K^T + pos) / sqrt(dh)
// grid (8 s-blocks of 64, 32 t-tiles, 64 (b,h)), block 128
// ---------------------------------------------------------------------------
__global__ __launch_bounds__(128)
void k_content(const unsigned short* __restrict__ qb,
               const unsigned short* __restrict__ kb,
               float* __restrict__ scores) {
    int lane = threadIdx.x & 31, wv = threadIdx.x >> 5;
    int bh = blockIdx.z, tt = blockIdx.y;
    int s0 = blockIdx.x * 64 + wv * 16;
    const unsigned short* abase = qb + (size_t)(bh * Tn + tt * 16) * DHn;
    const unsigned short* bbase = kb + (size_t)(bh * Tn + s0) * DHn;
    f32x8 c = {};
    #pragma unroll
    for (int k = 0; k < DHn; k += 32)
        c = wmma_bf16(load_a_bf16(abase + k, DHn, lane),
                      load_b_bf16(bbase + k, DHn, lane), c);

    int col = lane & 15, m0 = (lane >> 4) << 3;
    #pragma unroll
    for (int r = 0; r < 8; ++r) {
        size_t idx = ((size_t)bh * Tn + tt * 16 + m0 + r) * Tn + s0 + col;
        scores[idx] = (c[r] + scores[idx]) * 0.125f;   // 1/sqrt(64)
    }
}

// ---------------------------------------------------------------------------
// K4: row softmax, in place.  One wave32 per 512-wide row; shfl_xor reductions.
// grid 8192, block 128 (4 rows per block)
// ---------------------------------------------------------------------------
__global__ __launch_bounds__(128)
void k_softmax(float* __restrict__ scores) {
    int lane = threadIdx.x & 31, wv = threadIdx.x >> 5;
    size_t row = (size_t)blockIdx.x * 4 + wv;
    float* p = scores + row * Tn;
    float v[16];
    float mx = -3.4e38f;
    #pragma unroll
    for (int i = 0; i < 16; ++i) { v[i] = p[lane + i * 32]; mx = fmaxf(mx, v[i]); }
    #pragma unroll
    for (int off = 16; off; off >>= 1) mx = fmaxf(mx, __shfl_xor(mx, off, 32));
    float sum = 0.f;
    #pragma unroll
    for (int i = 0; i < 16; ++i) { v[i] = __expf(v[i] - mx); sum += v[i]; }
    #pragma unroll
    for (int off = 16; off; off >>= 1) sum += __shfl_xor(sum, off, 32);
    float inv = 1.0f / sum;
    #pragma unroll
    for (int i = 0; i < 16; ++i) p[lane + i * 32] = v[i] * inv;
}

// ---------------------------------------------------------------------------
// K5: context = P @ V   (P read f32 -> bf16 in-register; V pre-transposed)
// ctx stored [b][t][h*64+d] (bf16, contraction-contiguous for final GEMM)
// grid (32 t-tiles, 64 (b,h)), block 128 (wave w -> 16-col d-subtile)
// ---------------------------------------------------------------------------
__global__ __launch_bounds__(128)
void k_ctx(const float* __restrict__ scores,
           const unsigned short* __restrict__ vt,
           unsigned short* __restrict__ ctx) {
    int lane = threadIdx.x & 31, wv = threadIdx.x >> 5;
    int tt = blockIdx.x, bh = blockIdx.y;
    int d0 = wv * 16;
    const float* abase = scores + ((size_t)bh * Tn + tt * 16) * Tn;
    const unsigned short* bbase = vt + (size_t)(bh * DHn + d0) * Tn;
    f32x8 c = {};
    #pragma unroll 4
    for (int s = 0; s < Tn; s += 32)
        c = wmma_bf16(load_a_f32(abase + s, Tn, lane),
                      load_b_bf16(bbase + s, Tn, lane), c);

    int col = lane & 15, m0 = (lane >> 4) << 3;
    int b = bh >> 3, h = bh & 7;
    #pragma unroll
    for (int r = 0; r < 8; ++r) {
        int t = tt * 16 + m0 + r;
        ctx[((size_t)(b * Tn + t)) * Dm + h * DHn + d0 + col] =
            (unsigned short)f2bf(c[r]);
    }
}

// ---------------------------------------------------------------------------
// K6: output projection:  out = ctx @ Wo^T + bo   (f32 output)
// grid (256 m-tiles, 8 col-blocks), block 128
// ---------------------------------------------------------------------------
__global__ __launch_bounds__(128)
void k_oproj(const unsigned short* __restrict__ ctx,
             const unsigned short* __restrict__ wo,
             const float* __restrict__ bo,
             float* __restrict__ out) {
    int lane = threadIdx.x & 31, wv = threadIdx.x >> 5;
    int mt = blockIdx.x, cb = blockIdx.y;
    int c0 = cb * 64 + wv * 16;
    const unsigned short* abase = ctx + (size_t)mt * 16 * Dm;
    const unsigned short* bbase = wo + (size_t)c0 * Dm;
    f32x8 c = {};
    #pragma unroll 4
    for (int k = 0; k < Dm; k += 32)
        c = wmma_bf16(load_a_bf16(abase + k, Dm, lane),
                      load_b_bf16(bbase + k, Dm, lane), c);

    int col = lane & 15, m0 = (lane >> 4) << 3;
    float bvv = bo[c0 + col];
    #pragma unroll
    for (int r = 0; r < 8; ++r)
        out[(size_t)(mt * 16 + m0 + r) * Dm + c0 + col] = c[r] + bvv;
}

// ---------------------------------------------------------------------------
extern "C" void kernel_launch(void* const* d_in, const int* in_sizes, int n_in,
                              void* d_out, int out_size, void* d_ws, size_t ws_size,
                              hipStream_t stream) {
    const float* x   = (const float*)d_in[0];
    const float* rel = (const float*)d_in[1];
    const float* Wq  = (const float*)d_in[2];
    const float* bq  = (const float*)d_in[3];
    const float* Wk  = (const float*)d_in[4];
    const float* bk  = (const float*)d_in[5];
    const float* Wv  = (const float*)d_in[6];
    const float* bv  = (const float*)d_in[7];
    const float* Wo  = (const float*)d_in[8];
    const float* bo  = (const float*)d_in[9];
    float* out = (float*)d_out;

    // Workspace layout (bytes, all 256-aligned).  Total ~90.2 MB.
    char* ws = (char*)d_ws;
    unsigned short* xb  = (unsigned short*)(ws + 0);          //  4.19 MB bf16 x
    unsigned short* wqb = (unsigned short*)(ws + 4194304);    //  0.52 MB
    unsigned short* wkb = (unsigned short*)(ws + 4718592);
    unsigned short* wvb = (unsigned short*)(ws + 5242880);
    unsigned short* wob = (unsigned short*)(ws + 5767168);
    unsigned short* qb  = (unsigned short*)(ws + 6291456);    //  Q [b][h][t][d]
    unsigned short* kb  = (unsigned short*)(ws + 10485760);   //  K [b][h][t][d]
    unsigned short* vt  = (unsigned short*)(ws + 14680064);   //  V^T [b][h][d][t]
    float*          sc  = (float*)        (ws + 18874368);    //  scores 67 MB
    unsigned short* ctx = (unsigned short*)(ws + 85983232);   //  ctx bf16

    const int NX = Bn * Tn * Dm;      // 2,097,152
    const int NW = Dm * Dm;           //   262,144

    // K0: convert x and weights to bf16
    k_cvt_bf16<<<(NX + 255) / 256, 256, 0, stream>>>(x,  xb,  NX);
    k_cvt_bf16<<<(NW + 255) / 256, 256, 0, stream>>>(Wq, wqb, NW);
    k_cvt_bf16<<<(NW + 255) / 256, 256, 0, stream>>>(Wk, wkb, NW);
    k_cvt_bf16<<<(NW + 255) / 256, 256, 0, stream>>>(Wv, wvb, NW);
    k_cvt_bf16<<<(NW + 255) / 256, 256, 0, stream>>>(Wo, wob, NW);

    // K1: Q/K/V projections (V stored transposed for the PV GEMM)
    dim3 gProj(256, 8);
    k_qkv<<<gProj, 128, 0, stream>>>(xb, wqb, bq, qb, 0);
    k_qkv<<<gProj, 128, 0, stream>>>(xb, wkb, bk, kb, 1);
    k_qkv<<<gProj, 128, 0, stream>>>(xb, wvb, bv, vt, 2);

    // K2: position scores (single async-pipelined pass over 537 MB rel_emb)
    k_pos<<<dim3(Tn / TT, 8, NHn), 128, 0, stream>>>(qb, rel, sc);

    // K3: content scores + combine + scale
    k_content<<<dim3(8, 32, Bn * NHn), 128, 0, stream>>>(qb, kb, sc);

    // K4: in-place softmax over the s axis
    k_softmax<<<(Bn * NHn * Tn) / 4, 128, 0, stream>>>(sc);

    // K5: context = P @ V
    k_ctx<<<dim3(32, Bn * NHn), 128, 0, stream>>>(sc, vt, ctx);

    // K6: output projection
    k_oproj<<<dim3(256, 8), 128, 0, stream>>>(ctx, wob, bo, out);
}
